// MegatronMoEWrapper_19439021982129
// MI455X (gfx1250) — compile-verified, matrix operand order checked by reading, not run on hardware
//
#include <hip/hip_runtime.h>
#include <hip/hip_bf16.h>
#include <math.h>

typedef __bf16 bf16_t;
typedef __attribute__((ext_vector_type(16))) __bf16 v16bf;
typedef __attribute__((ext_vector_type(8)))  __bf16 v8bf;
typedef __attribute__((ext_vector_type(8)))  float  v8f;

#define E_ 8
#define D_ 1024
#define F_ 4096
#define N_ 2048

// native bf16 convert: lowers to v_cvt_pk_bf16_f32 pairs on gfx1250 (verified)
__device__ __forceinline__ bf16_t f2bf(float f) { return (bf16_t)f; }

__device__ __forceinline__ v8bf cvt8(float4 a, float4 b) {
  v8bf r;
  r[0]=f2bf(a.x); r[1]=f2bf(a.y); r[2]=f2bf(a.z); r[3]=f2bf(a.w);
  r[4]=f2bf(b.x); r[5]=f2bf(b.y); r[6]=f2bf(b.z); r[7]=f2bf(b.w);
  return r;
}

// gfx1250 async global->LDS copy (ASYNCcnt-tracked, bypasses VGPRs).
// INST_OFFSET is added to BOTH the LDS and global address (ISA §15.18).
__device__ __forceinline__ void async_copy32B(unsigned lds_off, const void* gptr) {
  unsigned long long ga = (unsigned long long)gptr;
  asm volatile("global_load_async_to_lds_b128 %0, %1, off\n\t"
               "global_load_async_to_lds_b128 %0, %1, off offset:16"
               :: "v"(lds_off), "v"(ga) : "memory");
}
__device__ __forceinline__ void wait_async0() {
  asm volatile("s_wait_asynccnt 0x0" ::: "memory");
}

// ---------------- init: zero output + expert counters ----------------
__global__ __launch_bounds__(256) void moe_init(float* __restrict__ out,
                                                int* __restrict__ counts) {
  int i = blockIdx.x * 256 + threadIdx.x;
  if (i < N_ * D_) out[i] = 0.f;
  if (blockIdx.x == 0 && threadIdx.x < E_) counts[threadIdx.x] = 0;
}

// ---------------- gating: logits -> top2 softmax -> coef + token lists ----------------
__global__ __launch_bounds__(256) void moe_gate(const float* __restrict__ x,
                                                const float* __restrict__ gw,
                                                float* __restrict__ coef,
                                                int* __restrict__ counts,
                                                int* __restrict__ tokidx) {
  const int n    = blockIdx.x;
  const int wave = threadIdx.x >> 5;   // one wave per expert (8 waves)
  const int lane = threadIdx.x & 31;
  __shared__ float logits[E_];

  const float* xr = x  + (long)n    * D_;
  const float* gr = gw + (long)wave * D_;
  float s = 0.f;
  #pragma unroll
  for (int d = lane; d < D_; d += 32) s += xr[d] * gr[d];
  #pragma unroll
  for (int off = 16; off; off >>= 1) s += __shfl_xor(s, off, 32);
  if (lane == 0) logits[wave] = s;
  __syncthreads();

  if (threadIdx.x == 0) {
    int   i0 = 0; float v0 = logits[0];
    for (int e = 1; e < E_; ++e) if (logits[e] > v0) { v0 = logits[e]; i0 = e; }
    int   i1 = -1; float v1 = -INFINITY;
    for (int e = 0; e < E_; ++e) if (e != i0 && logits[e] > v1) { v1 = logits[e]; i1 = e; }
    float w0 = 1.f / (1.f + __expf(v1 - v0));   // softmax over top-2
    float w1 = 1.f - w0;
    #pragma unroll
    for (int e = 0; e < E_; ++e) coef[n * E_ + e] = 0.f;
    coef[n * E_ + i0] = w0;
    coef[n * E_ + i1] = w1;
    int s0 = atomicAdd(&counts[i0], 1); tokidx[i0 * N_ + s0] = n;
    int s1 = atomicAdd(&counts[i1], 1); tokidx[i1 * N_ + s1] = n;
  }
}

// ---------------- x fp32 -> bf16 ----------------
__global__ __launch_bounds__(256) void cvt_bf16(const float* __restrict__ in,
                                                bf16_t* __restrict__ out, int n) {
  int i = blockIdx.x * 256 + threadIdx.x;
  if (i < n) out[i] = f2bf(in[i]);
}

// ---------------- fused expert GEMM (double-buffered, async A staging) ----------------
// MODE 0: A = xb gathered via tokidx (compacted slots), epilogue = GELU -> h (bf16)
// MODE 1: A = h (compacted), epilogue = out[tok] += coef * acc
template<int KDIM, int MODE>
__global__ __launch_bounds__(256) void moe_gemm(const bf16_t* __restrict__ A,
                                                const float*  __restrict__ B,   // [Ncols, KDIM] row-major
                                                bf16_t* __restrict__ Hout,
                                                float*  __restrict__ Out,
                                                const int*   __restrict__ counts,
                                                const int*   __restrict__ tokidx,
                                                const float* __restrict__ coef,
                                                int expert) {
  const int cnt = counts[expert];
  const int tile_m = blockIdx.y;      // token-slot tiles (compacted)
  const int tile_n = blockIdx.x;      // output-column tiles
  if (tile_m * 128 >= cnt) return;    // block-uniform early exit

  constexpr int BK   = 32;
  constexpr int LDST = 40;            // halves: 80B rows, 16B aligned, bank-skewed
  constexpr int BUF  = 128 * LDST;    // halves per buffer
  __shared__ bf16_t As[2 * BUF];
  __shared__ bf16_t Bs[2 * BUF];

  const int tid  = threadIdx.x;
  const int lane = tid & 31;
  const int wave = tid >> 5;
  const int wm   = wave & 3;          // 4 wave rows  (32 rows each)
  const int wn   = wave >> 2;         // 2 wave cols  (64 cols each)

  // ---- staging: each thread owns (row = tid>>1, 16-half segment = tid&1) ----
  const int arow = tid >> 1;
  const int aseg = (tid & 1) * 16;
  const int aslot = tile_m * 128 + arow;
  long asrc;
  if (MODE == 0) {
    int tok = (aslot < cnt) ? tokidx[expert * N_ + aslot] : 0;
    asrc = (long)tok * KDIM;
  } else {
    asrc = (long)((aslot < cnt) ? aslot : 0) * KDIM;
  }
  const bf16_t* aptr = A + asrc + aseg;
  // LDS byte offset for async dst: low 32 bits of flat shared address == LDS addr
  const unsigned aoff0 = (unsigned)(unsigned long long)(As + arow * LDST + aseg);
  const unsigned aoff1 = aoff0 + BUF * 2;   // +bytes to buffer 1

  const float* bptr = B + (long)(tile_n * 128 + arow) * KDIM + aseg;
  bf16_t* bdst0 = Bs + arow * LDST + aseg;

  v8f acc[2][4];
  #pragma unroll
  for (int i = 0; i < 2; ++i)
    #pragma unroll
    for (int j = 0; j < 4; ++j)
      acc[i][j] = (v8f){0.f,0.f,0.f,0.f,0.f,0.f,0.f,0.f};

  const int kb    = (lane >> 4) * 8;  // lanes<16: K {0..7,16..23}; lanes>=16: {8..15,24..31}
  const int mrow0 = wm * 32 + (lane & 15);
  const int ncol0 = wn * 64 + (lane & 15);

  constexpr int NK = KDIM / BK;
  float4 breg[4];

  // prologue: tile 0 in flight
  {
    const float4* g = (const float4*)(bptr);
    breg[0] = g[0]; breg[1] = g[1]; breg[2] = g[2]; breg[3] = g[3];
    async_copy32B(aoff0, aptr);
  }

  for (int k = 0; k < NK; ++k) {
    const int cur = k & 1;
    wait_async0();                      // A tile (cur) fully in LDS
    {                                   // B tile: fp32 regs -> bf16 LDS (cur)
      bf16_t* d = bdst0 + cur * BUF;
      *(v8bf*)d       = cvt8(breg[0], breg[1]);
      *(v8bf*)(d + 8) = cvt8(breg[2], breg[3]);
    }
    __syncthreads();                    // tile cur visible; all waves done with tile k-1

    if (k + 1 < NK) {                   // issue tile k+1 into the other buffer
      const int k1 = (k + 1) * BK;
      const float4* g = (const float4*)(bptr + k1);
      breg[0] = g[0]; breg[1] = g[1]; breg[2] = g[2]; breg[3] = g[3];
      async_copy32B((cur ? aoff0 : aoff1), aptr + k1);
      if (k + 2 < NK) __builtin_prefetch(bptr + (k + 2) * BK, 0, 3);  // near-cache prefetch
    }

    // ---- compute tile cur: load ALL fragments first, then 8 back-to-back WMMAs ----
    const bf16_t* Asb = As + cur * BUF;
    const bf16_t* Bsb = Bs + cur * BUF;
    const bf16_t* ar0 = Asb + (mrow0)      * LDST + kb;
    const bf16_t* ar1 = Asb + (mrow0 + 16) * LDST + kb;
    v8bf a0lo = *(const v8bf*)(ar0), a0hi = *(const v8bf*)(ar0 + 16);
    v8bf a1lo = *(const v8bf*)(ar1), a1hi = *(const v8bf*)(ar1 + 16);

    v8bf blo[4], bhi[4];
    #pragma unroll
    for (int nt = 0; nt < 4; ++nt) {
      const bf16_t* br = Bsb + (ncol0 + nt * 16) * LDST + kb;
      blo[nt] = *(const v8bf*)(br);
      bhi[nt] = *(const v8bf*)(br + 16);
    }

    v16bf af0 = __builtin_shufflevector(a0lo, a0hi, 0,1,2,3,4,5,6,7,8,9,10,11,12,13,14,15);
    v16bf af1 = __builtin_shufflevector(a1lo, a1hi, 0,1,2,3,4,5,6,7,8,9,10,11,12,13,14,15);

    #pragma unroll
    for (int nt = 0; nt < 4; ++nt) {
      v16bf bfv = __builtin_shufflevector(blo[nt], bhi[nt], 0,1,2,3,4,5,6,7,8,9,10,11,12,13,14,15);
      acc[0][nt] = __builtin_amdgcn_wmma_f32_16x16x32_bf16(false, af0, false, bfv,
                                                           (short)0, acc[0][nt], false, false);
      acc[1][nt] = __builtin_amdgcn_wmma_f32_16x16x32_bf16(false, af1, false, bfv,
                                                           (short)0, acc[1][nt], false, false);
    }
  }

  // ---- epilogue ----
  // C/D layout: VGPR r, lane<16 -> M=r, N=lane; lane>=16 -> M=8+r, N=lane-16
  const int rofs = (lane < 16) ? 0 : 8;
  #pragma unroll
  for (int mt = 0; mt < 2; ++mt) {
    const int rbase = tile_m * 128 + wm * 32 + mt * 16 + rofs;
    #pragma unroll
    for (int nt = 0; nt < 4; ++nt) {
      const int col = tile_n * 128 + wn * 64 + nt * 16 + (lane & 15);
      #pragma unroll
      for (int r = 0; r < 8; ++r) {
        const int row = rbase + r;
        if (row < cnt) {
          float v = acc[mt][nt][r];
          if (MODE == 0) {
            float g = 0.5f * v * (1.f + erff(v * 0.70710678118654752f));  // exact GELU
            Hout[(long)row * F_ + col] = f2bf(g);
          } else {
            int   tok = tokidx[expert * N_ + row];
            float c   = coef[tok * E_ + expert];
            float* o  = Out + (long)tok * D_ + col;   // unique writer this launch
            *o += c * v;
          }
        }
      }
    }
  }
}

// ---------------- host launch ----------------
extern "C" void kernel_launch(void* const* d_in, const int* in_sizes, int n_in,
                              void* d_out, int out_size, void* d_ws, size_t ws_size,
                              hipStream_t stream) {
  const float* x      = (const float*)d_in[0];
  const float* gate_w = (const float*)d_in[1];
  const float* w1     = (const float*)d_in[2];
  const float* w2     = (const float*)d_in[3];
  float* out = (float*)d_out;

  char* ws = (char*)d_ws;
  int*    counts = (int*)ws;                                   // 32 B
  int*    tokidx = (int*)(ws + 64);                            // E*N*4  = 64 KB
  float*  coef   = (float*)(ws + 64 + E_ * N_ * 4);            // N*E*4  = 64 KB
  bf16_t* xb     = (bf16_t*)(ws + 64 + E_ * N_ * 4 + N_ * E_ * 4);          // 4 MB
  bf16_t* h      = (bf16_t*)((char*)xb + (size_t)N_ * D_ * 2);              // 16 MB

  moe_init<<<(N_ * D_ + 255) / 256, 256, 0, stream>>>(out, counts);
  moe_gate<<<N_, 256, 0, stream>>>(x, gate_w, coef, counts, tokidx);
  cvt_bf16<<<(N_ * D_ + 255) / 256, 256, 0, stream>>>(x, xb, N_ * D_);

  for (int e = 0; e < E_; ++e) {
    moe_gemm<D_, 0><<<dim3(F_ / 128, N_ / 128), 256, 0, stream>>>(
        xb, w1 + (long)e * F_ * D_, h, nullptr, counts, tokidx, coef, e);
    moe_gemm<F_, 1><<<dim3(D_ / 128, N_ / 128), 256, 0, stream>>>(
        h, w2 + (long)e * D_ * F_, nullptr, out, counts, tokidx, coef, e);
  }
}
